// PointNet2_81329500717170
// MI455X (gfx1250) — matrix operand order repaired and checked by reading, not compiled, not run on hardware
//
#include <hip/hip_runtime.h>
#include <hip/hip_bf16.h>
#include <hip/hip_fp16.h>
#include <stdint.h>

typedef __attribute__((ext_vector_type(16))) _Float16 v16h;
typedef __attribute__((ext_vector_type(8)))  float    v8f;
typedef __attribute__((ext_vector_type(4)))  int      v4i;

#define BN_EPS 1e-5f
#define BATCH 64

#if defined(__has_builtin)
#  if __has_builtin(__builtin_amdgcn_global_load_async_to_lds_b128)
#    define HAVE_ASYNC_LDS 1
#  endif
#  if __has_builtin(__builtin_amdgcn_s_wait_asynccnt)
#    define HAVE_WAIT_ASYNC_BUILTIN 1
#  endif
#endif

// ---------------------------------------------------------------------------
// Stage a [16-byte-multiple] block of global memory into LDS.
// Preferred path: GLOBAL_LOAD_ASYNC_TO_LDS_B128 (ASYNCcnt, no VGPR round trip).
// Fallback: b128 loads + ds stores.  Caller must __syncthreads() after.
// ---------------------------------------------------------------------------
__device__ __forceinline__ void stage_to_lds(const float* __restrict__ src,
                                             float* dst_lds, int nbytes,
                                             int tid, int nthreads) {
#ifdef HAVE_ASYNC_LDS
  __attribute__((address_space(1))) v4i* g =
      (__attribute__((address_space(1))) v4i*)src;
  __attribute__((address_space(3))) v4i* l =
      (__attribute__((address_space(3))) v4i*)dst_lds;
  int n16 = nbytes >> 4;
  for (int i = tid; i < n16; i += nthreads)
    __builtin_amdgcn_global_load_async_to_lds_b128(g + i, l + i, 0, 0);
#  ifdef HAVE_WAIT_ASYNC_BUILTIN
  __builtin_amdgcn_s_wait_asynccnt(0);
#  else
  asm volatile("s_wait_asynccnt 0x0" ::: "memory");
#  endif
#else
  const float4* s4 = (const float4*)src;
  float4* d4 = (float4*)dst_lds;
  int n4 = nbytes >> 4;
  for (int i = tid; i < n4; i += nthreads) d4[i] = s4[i];
#endif
}

// ---------------------------------------------------------------------------
// Weight prep: f32 W (Cout x Cin) -> f16 padded (Cout x Cpad), fold BN into
// per-channel scale/shift:  y = a*x + b,  a = g*rsqrt(v+eps),  b = be - a*rm
// ---------------------------------------------------------------------------
__global__ void prep_weights(const float* __restrict__ W, const float* __restrict__ g,
                             const float* __restrict__ be, const float* __restrict__ rm,
                             const float* __restrict__ rv,
                             _Float16* __restrict__ Wh, float* __restrict__ bnA,
                             float* __restrict__ bnB, int Cout, int Cin, int Cpad) {
  int stride = gridDim.x * blockDim.x;
  int t0 = blockIdx.x * blockDim.x + threadIdx.x;
  int total = Cout * Cpad;
  for (int t = t0; t < total; t += stride) {
    int c = t / Cpad, k = t % Cpad;
    float v = (k < Cin) ? W[c * Cin + k] : 0.f;
    Wh[t] = (_Float16)v;
  }
  for (int c = t0; c < Cout; c += stride) {
    float a = g[c] * rsqrtf(rv[c] + BN_EPS);
    bnA[c] = a;
    bnB[c] = be[c] - a * rm[c];
  }
}

// ---------------------------------------------------------------------------
// Farthest point sampling: one workgroup per batch. Cloud staged in LDS
// (async copy), running min-distance kept in registers, LDS tree argmax per
// step. Also emits new_xyz (gathered centers) directly.
// ---------------------------------------------------------------------------
#define FPS_BT 256
#define FPS_PT 16   // 4096 / 256
__global__ void fps_kernel(const float* __restrict__ xyz, int N, int npoint,
                           int* __restrict__ out_idx, float* __restrict__ new_xyz) {
  __shared__ __align__(16) float s_xyz[4096 * 3];
  __shared__ float s_val[FPS_BT];
  __shared__ int   s_ind[FPS_BT];
  __shared__ float s_pt[3];
  const int b = blockIdx.x;
  const int tid = threadIdx.x;
  const float* xb = xyz + (size_t)b * N * 3;
  stage_to_lds(xb, s_xyz, N * 12, tid, FPS_BT);
  __syncthreads();

  float dist[FPS_PT];
#pragma unroll
  for (int i = 0; i < FPS_PT; i++) dist[i] = 1e10f;

  if (tid == 0) {
    out_idx[(size_t)b * npoint] = 0;
    new_xyz[(size_t)b * npoint * 3 + 0] = s_xyz[0];
    new_xyz[(size_t)b * npoint * 3 + 1] = s_xyz[1];
    new_xyz[(size_t)b * npoint * 3 + 2] = s_xyz[2];
    s_pt[0] = s_xyz[0]; s_pt[1] = s_xyz[1]; s_pt[2] = s_xyz[2];
  }
  __syncthreads();

  for (int it = 1; it < npoint; it++) {
    const float px = s_pt[0], py = s_pt[1], pz = s_pt[2];
    float bv = -1.f; int bi = 0;
#pragma unroll
    for (int i = 0; i < FPS_PT; i++) {
      int gi = tid + i * FPS_BT;
      if (gi < N) {
        float dx = s_xyz[gi * 3 + 0] - px;
        float dy = s_xyz[gi * 3 + 1] - py;
        float dz = s_xyz[gi * 3 + 2] - pz;
        float d = dx * dx + dy * dy + dz * dz;
        float nd = fminf(dist[i], d);
        dist[i] = nd;
        if (nd > bv) { bv = nd; bi = gi; }
      }
    }
    s_val[tid] = bv; s_ind[tid] = bi;
    __syncthreads();
    for (int off = FPS_BT / 2; off > 0; off >>= 1) {
      if (tid < off) {
        float ov = s_val[tid + off]; int oi = s_ind[tid + off];
        float cv = s_val[tid];       int ci = s_ind[tid];
        if (ov > cv || (ov == cv && oi < ci)) { s_val[tid] = ov; s_ind[tid] = oi; }
      }
      __syncthreads();
    }
    if (tid == 0) {
      int sel = s_ind[0];
      out_idx[(size_t)b * npoint + it] = sel;
      float x = s_xyz[sel * 3 + 0], y = s_xyz[sel * 3 + 1], z = s_xyz[sel * 3 + 2];
      new_xyz[((size_t)b * npoint + it) * 3 + 0] = x;
      new_xyz[((size_t)b * npoint + it) * 3 + 1] = y;
      new_xyz[((size_t)b * npoint + it) * 3 + 2] = z;
      s_pt[0] = x; s_pt[1] = y; s_pt[2] = z;
    }
    __syncthreads();
  }
}

// ---------------------------------------------------------------------------
// Ball query: one workgroup per batch; cloud staged in LDS (async); one
// thread per center scans indices ascending (== reference sort-then-truncate).
// ---------------------------------------------------------------------------
__global__ void ballq_kernel(const float* __restrict__ new_xyz,
                             const float* __restrict__ xyz,
                             int N, int S, float r2, int nsample,
                             int* __restrict__ out_idx) {
  __shared__ __align__(16) float s_xyz[4096 * 3];
  const int b = blockIdx.x;
  const float* xb = xyz + (size_t)b * N * 3;
  stage_to_lds(xb, s_xyz, N * 12, threadIdx.x, blockDim.x);
  __syncthreads();
  for (int s = threadIdx.x; s < S; s += blockDim.x) {
    float cx = new_xyz[((size_t)b * S + s) * 3 + 0];
    float cy = new_xyz[((size_t)b * S + s) * 3 + 1];
    float cz = new_xyz[((size_t)b * S + s) * 3 + 2];
    int* o = out_idx + ((size_t)b * S + s) * nsample;
    int cnt = 0, first = 0;
    for (int j = 0; j < N && cnt < nsample; j++) {
      float dx = s_xyz[j * 3 + 0] - cx;
      float dy = s_xyz[j * 3 + 1] - cy;
      float dz = s_xyz[j * 3 + 2] - cz;
      if (dx * dx + dy * dy + dz * dz < r2) {
        if (cnt == 0) first = j;
        o[cnt++] = j;
      }
    }
    for (int k = cnt; k < nsample; k++) o[k] = first;
  }
}

// ---------------------------------------------------------------------------
// Set-abstraction MLP: one workgroup (4 wave32s) per (batch, center).
//   1) gather grouped f16 input X (Cpad x NS) into LDS, [j][c] layout
//   2) Y1 = relu(bn(W0 * X))  via v_wmma_f32_16x16x32_f16, stored f16 in LDS
//   3) Y2 = relu(bn(W1 * Y1)) via WMMA, neighbor-max in registers (shfl_xor)
// A-fragment uses the same contiguous K order as B; B-fragment: lanes 0-15 K=0-15,
// lanes 16-31 K=16-31, element e = K offset. C/D: lane n = L%16, m = r+8*(L/16).
// ---------------------------------------------------------------------------
template <int CF, int CPAD, int C1, int C2, int NS, int FEAT_FROM_PC, int NSRC>
__global__ void sa_mlp_kernel(const float* __restrict__ xyz,     // (B, NSRC, 3)
                              const float* __restrict__ feats,   // (B, CF, NSRC) or pc
                              const float* __restrict__ new_xyz, // (B, S, 3)
                              const int* __restrict__ idx,       // (B, S, NS)
                              const _Float16* __restrict__ W0,   // (C1, CPAD)
                              const float* __restrict__ bnA0, const float* __restrict__ bnB0,
                              const _Float16* __restrict__ W1,   // (C2, C1)
                              const float* __restrict__ bnA1, const float* __restrict__ bnB1,
                              float* __restrict__ out,           // (B, C2, S)
                              int S) {
  constexpr int NW = 4;   // waves per block (blockDim.x == 128)
  __shared__ __align__(32) _Float16 sX[NS * CPAD];
  __shared__ __align__(32) _Float16 sY[NS * C1];
  __shared__ int   s_pi[NS];
  __shared__ float s_ctr[3];

  const int b = blockIdx.y;
  const int s = blockIdx.x;
  const int tid = threadIdx.x;

  if (tid < NS) s_pi[tid] = idx[((size_t)b * S + s) * NS + tid];
  if (tid < 3)  s_ctr[tid] = new_xyz[((size_t)b * S + s) * 3 + tid];
  __syncthreads();

  // ---- gather grouped input: sX[j*CPAD + c]
  for (int t = tid; t < NS * CPAD; t += 128) {
    int j = t / CPAD, c = t % CPAD;
    int p = s_pi[j];
    float v;
    if (c < 3) {
      v = xyz[((size_t)b * NSRC + p) * 3 + c] - s_ctr[c];
    } else if (c < 3 + CF) {
      if (FEAT_FROM_PC) v = feats[((size_t)b * NSRC + p) * 3 + ((c - 3) % 3)];
      else              v = feats[((size_t)b * CF + (c - 3)) * NSRC + p];
    } else {
      v = 0.f;
    }
    sX[t] = (_Float16)v;
  }
  __syncthreads();

  const int wave = tid >> 5;
  const int lane = tid & 31;
  const int nl = lane & 15;   // column within tile / m-row for A
  const int hl = lane >> 4;   // lane half

  // ---- layer 0: Y1 = relu(bn(W0 * X)) -> sY[j*C1 + c] (f16)
  constexpr int MT0 = C1 / 16, NT0 = NS / 16, T0 = MT0 * NT0;
  for (int tile = wave; tile < T0; tile += NW) {
    int mt = tile / NT0, nt = tile % NT0;
    v8f acc = {};
    for (int kt = 0; kt < CPAD / 32; kt++) {
      v16h a, bm;
      const uint32_t* wrow =
          (const uint32_t*)(W0 + (size_t)(mt * 16 + nl) * CPAD + kt * 32);
      uint32_t* ap = (uint32_t*)&a;
#pragma unroll
      for (int v = 0; v < 8; v++) {
        int K = hl * 16 + v * 2;
        ap[v] = wrow[K >> 1];
      }
      bm = *(const v16h*)(sX + (size_t)(nt * 16 + nl) * CPAD + kt * 32 + hl * 16);
      acc = __builtin_amdgcn_wmma_f32_16x16x32_f16(false, a, false, bm,
                                                   (short)0, acc, false, false);
    }
#pragma unroll
    for (int r = 0; r < 8; r++) {
      int c = mt * 16 + r + 8 * hl;
      int j = nt * 16 + nl;
      float y = fmaxf(0.f, bnA0[c] * acc[r] + bnB0[c]);
      sY[j * C1 + c] = (_Float16)y;
    }
  }
  __syncthreads();

  // ---- layer 1 + neighbor max-pool
  constexpr int MT1 = C2 / 16, NT1 = NS / 16;
  float* ob = out + (size_t)b * C2 * S + s;
  for (int mt = wave; mt < MT1; mt += NW) {
    float vmax[8];
#pragma unroll
    for (int r = 0; r < 8; r++) vmax[r] = -1e30f;
    for (int nt = 0; nt < NT1; nt++) {
      v8f acc = {};
      for (int kt = 0; kt < C1 / 32; kt++) {
        v16h a, bm;
        const uint32_t* wrow =
            (const uint32_t*)(W1 + (size_t)(mt * 16 + nl) * C1 + kt * 32);
        uint32_t* ap = (uint32_t*)&a;
#pragma unroll
        for (int v = 0; v < 8; v++) {
          int K = hl * 16 + v * 2;
          ap[v] = wrow[K >> 1];
        }
        bm = *(const v16h*)(sY + (size_t)(nt * 16 + nl) * C1 + kt * 32 + hl * 16);
        acc = __builtin_amdgcn_wmma_f32_16x16x32_f16(false, a, false, bm,
                                                     (short)0, acc, false, false);
      }
#pragma unroll
      for (int r = 0; r < 8; r++) {
        int c = mt * 16 + r + 8 * hl;
        float y = fmaxf(0.f, bnA1[c] * acc[r] + bnB1[c]);
        vmax[r] = fmaxf(vmax[r], y);
      }
    }
    // reduce max over 16 columns (lanes 0-15 and 16-31 hold distinct m rows)
#pragma unroll
    for (int r = 0; r < 8; r++) {
      float v = vmax[r];
      for (int m = 8; m >= 1; m >>= 1) v = fmaxf(v, __shfl_xor(v, m, 32));
      if (nl == 0) {
        int c = mt * 16 + r + 8 * hl;
        ob[(size_t)c * S] = v;
      }
    }
  }
}

// ---------------------------------------------------------------------------
// Host orchestration
// ---------------------------------------------------------------------------
extern "C" void kernel_launch(void* const* d_in, const int* in_sizes, int n_in,
                              void* d_out, int out_size, void* d_ws, size_t ws_size,
                              hipStream_t stream) {
  (void)in_sizes; (void)n_in; (void)out_size; (void)ws_size;
  const float* pc = (const float*)d_in[0];
  const float *Wf[6], *Gf[6], *Bef[6], *Rmf[6], *Rvf[6];
  for (int l = 0; l < 6; l++) {
    Wf[l]  = (const float*)d_in[1 + 5 * l + 0];
    Gf[l]  = (const float*)d_in[1 + 5 * l + 1];
    Bef[l] = (const float*)d_in[1 + 5 * l + 2];
    Rmf[l] = (const float*)d_in[1 + 5 * l + 3];
    Rvf[l] = (const float*)d_in[1 + 5 * l + 4];
  }

  // ---- workspace carving
  char* ws = (char*)d_ws;
  size_t off = 0;
  auto carve = [&](size_t bytes) -> void* {
    off = (off + 255) & ~(size_t)255;
    void* p = ws + off;
    off += bytes;
    return p;
  };
  int*   fps1   = (int*)  carve((size_t)BATCH * 128 * 4);
  float* xyz1   = (float*)carve((size_t)BATCH * 128 * 3 * 4);
  int*   ball1  = (int*)  carve((size_t)BATCH * 128 * 32 * 4);
  float* feats1 = (float*)carve((size_t)BATCH * 64 * 128 * 4);
  int*   fps2   = (int*)  carve((size_t)BATCH * 64 * 4);
  float* xyz2   = (float*)carve((size_t)BATCH * 64 * 3 * 4);
  int*   ball2  = (int*)  carve((size_t)BATCH * 64 * 32 * 4);
  float* feats2 = (float*)carve((size_t)BATCH * 128 * 64 * 4);
  int*   fps3   = (int*)  carve((size_t)BATCH * 32 * 4);
  float* xyz3   = (float*)carve((size_t)BATCH * 32 * 3 * 4);
  int*   ball3  = (int*)  carve((size_t)BATCH * 32 * 16 * 4);

  static const int Cin[6]  = {9, 32, 67, 128, 131, 256};
  static const int Cout[6] = {32, 64, 128, 128, 256, 256};
  static const int Cpad[6] = {32, 32, 96, 128, 160, 256};
  _Float16* Wh[6];
  float *bnA[6], *bnB[6];
  for (int l = 0; l < 6; l++) {
    Wh[l]  = (_Float16*)carve((size_t)Cout[l] * Cpad[l] * 2);
    bnA[l] = (float*)carve((size_t)Cout[l] * 4);
    bnB[l] = (float*)carve((size_t)Cout[l] * 4);
  }

  // ---- weight conversion + BN folding
  for (int l = 0; l < 6; l++) {
    prep_weights<<<dim3(16), dim3(256), 0, stream>>>(
        Wf[l], Gf[l], Bef[l], Rmf[l], Rvf[l], Wh[l], bnA[l], bnB[l],
        Cout[l], Cin[l], Cpad[l]);
  }

  // ---- SA level 1: N=4096 -> S=128, nsample=32, r=0.02; Cin=9 (3 + 6 from pc||pc)
  fps_kernel<<<dim3(BATCH), dim3(FPS_BT), 0, stream>>>(pc, 4096, 128, fps1, xyz1);
  ballq_kernel<<<dim3(BATCH), dim3(128), 0, stream>>>(xyz1, pc, 4096, 128,
                                                      0.02f * 0.02f, 32, ball1);
  sa_mlp_kernel<6, 32, 32, 64, 32, 1, 4096>
      <<<dim3(128, BATCH), dim3(128), 0, stream>>>(
          pc, pc, xyz1, ball1, Wh[0], bnA[0], bnB[0], Wh[1], bnA[1], bnB[1],
          feats1, 128);

  // ---- SA level 2: N=128 -> S=64, nsample=32, r=0.04; Cin=67 -> pad 96
  fps_kernel<<<dim3(BATCH), dim3(FPS_BT), 0, stream>>>(xyz1, 128, 64, fps2, xyz2);
  ballq_kernel<<<dim3(BATCH), dim3(128), 0, stream>>>(xyz2, xyz1, 128, 64,
                                                      0.04f * 0.04f, 32, ball2);
  sa_mlp_kernel<64, 96, 128, 128, 32, 0, 128>
      <<<dim3(64, BATCH), dim3(128), 0, stream>>>(
          xyz1, feats1, xyz2, ball2, Wh[2], bnA[2], bnB[2], Wh[3], bnA[3], bnB[3],
          feats2, 64);

  // ---- SA level 3: N=64 -> S=32, nsample=16, r=0.08; Cin=131 -> pad 160
  fps_kernel<<<dim3(BATCH), dim3(FPS_BT), 0, stream>>>(xyz2, 64, 32, fps3, xyz3);
  ballq_kernel<<<dim3(BATCH), dim3(128), 0, stream>>>(xyz3, xyz2, 64, 32,
                                                      0.08f * 0.08f, 16, ball3);
  sa_mlp_kernel<128, 160, 256, 256, 16, 0, 64>
      <<<dim3(32, BATCH), dim3(128), 0, stream>>>(
          xyz2, feats2, xyz3, ball3, Wh[4], bnA[4], bnB[4], Wh[5], bnA[5], bnB[5],
          (float*)d_out, 32);
}
